// FuzzyRGCN_86053964742975
// MI455X (gfx1250) — matrix operand order repaired and checked by reading, not compiled, not run on hardware
//
#include <hip/hip_runtime.h>
#include <hip/hip_bf16.h>
#include <cstdint>
#include <cstddef>

// ---------------------------------------------------------------------------
// FuzzyRGCN on MI455X (gfx1250): f16-WMMA projection + L2-resident f16 hp
// hp[n][rq][o] kept in f16 (102.4 MB -> fits 192 MB L2); accum in f32.
// ---------------------------------------------------------------------------

typedef __attribute__((ext_vector_type(16))) _Float16 v16h;
typedef __attribute__((ext_vector_type(8)))  _Float16 v8h;
typedef __attribute__((ext_vector_type(8)))  float    v8f;
typedef __attribute__((ext_vector_type(4)))  float    v4f;

#define FIN   32
#define FOUT  32
#define NREL  8
#define NQ    4
#define RQ    32   // NREL*NQ

// -------------------- tiny utility kernels --------------------

__global__ void rgcn_zero_f32(float* __restrict__ p, int n) {
  int i = blockIdx.x * blockDim.x + threadIdx.x;
  if (i < n) p[i] = 0.0f;
}

__global__ void rgcn_cvt_f16(const float* __restrict__ in, _Float16* __restrict__ out, int n) {
  int i = blockIdx.x * blockDim.x + threadIdx.x;
  if (i < n) out[i] = (_Float16)in[i];
}

__global__ void rgcn_tanh_cvt(const float* __restrict__ in, _Float16* __restrict__ out, int n) {
  int i = blockIdx.x * blockDim.x + threadIdx.x;
  if (i < n) out[i] = (_Float16)tanhf(in[i]);
}

// Pack W [RQ][32][32] f32 (row-major, K x O per rq) into the per-lane B-operand
// layout for v_wmma_f32_16x16x32_f16:
//   B is 32x16 per (rq, ntile). Lane L holds column o = nt*16 + (L&15);
//   v16h element j of lane L is W[k][o] with k = 16*(j>>3) + 8*(L>>4) + (j&7).
// Wp flat index: ((rq*2 + nt)*32 + lane)*16 + j
__global__ void rgcn_pack_w(const float* __restrict__ W, _Float16* __restrict__ Wp) {
  int i = blockIdx.x * blockDim.x + threadIdx.x;
  if (i >= RQ * 2 * 32 * 16) return;
  int j    = i & 15;
  int lane = (i >> 4) & 31;
  int nt   = (i >> 9) & 1;
  int rq   = i >> 10;
  int k    = ((j >> 3) << 4) + ((lane >> 4) << 3) + (j & 7);   // 16*(j/8) + 8*(lane/16) + j%8
  int o    = (nt << 4) + (lane & 15);
  Wp[i] = (_Float16)W[((size_t)rq * FIN + k) * FOUT + o];
}

// -------------------- WMMA projection: hp = A @ W --------------------
// One wave per (rq), grid-strided over 16-node tiles, 2-deep software pipeline.
// A (16x32 f16): lane L (h=L/16, m=L%16) holds row m, K in [8h,8h+8) and
// [16+8h, 16+8h+8)  -> two contiguous 16B loads from the row-major f16 row.
// D (16x16 f32): lane col = L%16, VGPR v -> row v + 8h.
__global__ void rgcn_project_wmma(const _Float16* __restrict__ Ah,   // [N][32] f16
                                  const _Float16* __restrict__ Wp,   // packed B operands
                                  _Float16* __restrict__ hp,         // [N][RQ][32] f16
                                  int nNodes) {
  const int lane  = threadIdx.x & 31;
  const int wpb   = blockDim.x >> 5;
  const int wave  = blockIdx.x * wpb + (threadIdx.x >> 5);
  const int rq    = wave & (RQ - 1);
  const int tile0 = wave >> 5;                       // starting node-tile
  const int tstep = (gridDim.x * wpb) >> 5;          // tiles strided per rq
  const int nTiles = (nNodes + 15) >> 4;

  const int hh  = lane >> 4;      // half-wave selector
  const int r16 = lane & 15;      // A-row within tile / D-column

  // B operands for both 16-column halves: 32B per lane, held across all tiles.
  const v16h* wp = (const v16h*)Wp;
  const v16h b0 = wp[(rq * 2 + 0) * 32 + lane];
  const v16h b1 = wp[(rq * 2 + 1) * 32 + lane];

  int tile = tile0;
  if (tile >= nTiles) return;

  // prologue loads
  int arow0 = tile * 16 + r16;
  if (arow0 >= nNodes) arow0 = nNodes - 1;
  v8h lo = *(const v8h*)(Ah + (size_t)arow0 * FIN + 8 * hh);
  v8h hi = *(const v8h*)(Ah + (size_t)arow0 * FIN + 16 + 8 * hh);

  while (true) {
    const int next = tile + tstep;
    const bool hasNext = next < nTiles;

    v16h a;
#pragma unroll
    for (int i = 0; i < 8; ++i) { a[i] = lo[i]; a[i + 8] = hi[i]; }

    // issue next tile's A loads before the WMMAs (hide VMEM latency)
    if (hasNext) {
      int arowN = next * 16 + r16;
      if (arowN >= nNodes) arowN = nNodes - 1;
      lo = *(const v8h*)(Ah + (size_t)arowN * FIN + 8 * hh);
      hi = *(const v8h*)(Ah + (size_t)arowN * FIN + 16 + 8 * hh);
    }

    v8f c0 = {};
    v8f c1 = {};
    // 8 args: (neg_a, A, neg_b, B, c_mod, C, reuse_a, reuse_b)
    c0 = __builtin_amdgcn_wmma_f32_16x16x32_f16(false, a, false, b0, (short)0, c0, false, false);
    c1 = __builtin_amdgcn_wmma_f32_16x16x32_f16(false, a, false, b1, (short)0, c1, false, false);

    // D store: lane writes col r16 (c0) and col 16+r16 (c1) of rows v + 8*hh.
    _Float16* dbase = hp + (((size_t)(tile * 16 + 8 * hh) * RQ + rq) * FOUT) + r16;
    if (tile * 16 + 16 <= nNodes) {
      // full tile: branch-free, row stride RQ*FOUT (2048B) fits the imm offset
#pragma unroll
      for (int v = 0; v < 8; ++v) {
        dbase[(size_t)v * (RQ * FOUT)]      = (_Float16)c0[v];
        dbase[(size_t)v * (RQ * FOUT) + 16] = (_Float16)c1[v];
      }
    } else {
      // tail tile: guard per row
#pragma unroll
      for (int v = 0; v < 8; ++v) {
        if (tile * 16 + 8 * hh + v < nNodes) {
          dbase[(size_t)v * (RQ * FOUT)]      = (_Float16)c0[v];
          dbase[(size_t)v * (RQ * FOUT) + 16] = (_Float16)c1[v];
        }
      }
    }

    if (!hasNext) break;
    tile = next;
  }
}

// -------------------- per-edge combine + scatter-add --------------------
// One wave per edge (grid-strided); lane = output channel.
// msg[o] = sum_q tv[e,q] * (hp[src,rel,q,o] + b[rel,q,o]); atomicAdd into dst row.
__global__ void rgcn_edge_scatter(const _Float16* __restrict__ hp,   // [N][RQ][32] f16
                                  const float* __restrict__ bias,    // [NREL][NQ][32] f32
                                  const float* __restrict__ tv,      // [E][NQ] f32
                                  const int* __restrict__ src,
                                  const int* __restrict__ dst,
                                  const int* __restrict__ et,
                                  float* __restrict__ out,           // [N][32] f32 accum
                                  int nE) {
  const int lane = threadIdx.x & 31;
  const int wpb  = blockDim.x >> 5;
  const int wave = blockIdx.x * wpb + (threadIdx.x >> 5);
  const int nw   = gridDim.x * wpb;

  for (int e = wave; e < nE; e += nw) {
    const int s = src[e];
    const int d = dst[e];
    const int r = et[e];

    if (e + nw < nE) {                      // pull next edge's indices toward L2
      __builtin_prefetch(&src[e + nw], 0, 1);
      __builtin_prefetch(&tv[(size_t)(e + nw) * NQ], 0, 1);
    }

    const _Float16* hrow = hp + ((size_t)s * RQ + (size_t)r * NQ) * FOUT + lane;
    const float*    brow = bias + ((size_t)r * NQ) * FOUT + lane;
    const v4f t = *(const v4f*)(tv + (size_t)e * NQ);   // one b128 load

    float acc;
    acc  = t.x * ((float)hrow[0 * FOUT] + brow[0 * FOUT]);
    acc += t.y * ((float)hrow[1 * FOUT] + brow[1 * FOUT]);
    acc += t.z * ((float)hrow[2 * FOUT] + brow[2 * FOUT]);
    acc += t.w * ((float)hrow[3 * FOUT] + brow[3 * FOUT]);

    atomicAdd(out + (size_t)d * FOUT + lane, acc);   // global_atomic_add_f32
  }
}

// -------------------- launcher --------------------

static inline size_t align256(size_t x) { return (x + 255) & ~(size_t)255; }

extern "C" void kernel_launch(void* const* d_in, const int* in_sizes, int n_in,
                              void* d_out, int out_size, void* d_ws, size_t ws_size,
                              hipStream_t stream) {
  const float* feat = (const float*)d_in[0];
  const float* tv   = (const float*)d_in[1];
  const float* W1   = (const float*)d_in[2];
  const float* b1   = (const float*)d_in[3];
  const float* W2   = (const float*)d_in[4];
  const float* b2   = (const float*)d_in[5];
  const int*   src  = (const int*)d_in[6];
  const int*   dst  = (const int*)d_in[7];
  const int*   et   = (const int*)d_in[8];

  const int N = in_sizes[0] / FIN;   // 50000
  const int E = in_sizes[6];         // 800000
  const int NF = N * FIN;            // node feature count

  // workspace carve-up (~115 MB total)
  size_t off = 0;
  _Float16* feat_h = (_Float16*)((char*)d_ws + off); off = align256(off + (size_t)NF * 2);
  _Float16* xh     = (_Float16*)((char*)d_ws + off); off = align256(off + (size_t)NF * 2);
  _Float16* Wp1    = (_Float16*)((char*)d_ws + off); off = align256(off + (size_t)RQ * 2 * 32 * 16 * 2);
  _Float16* Wp2    = (_Float16*)((char*)d_ws + off); off = align256(off + (size_t)RQ * 2 * 32 * 16 * 2);
  float*    x      = (float*)((char*)d_ws + off);    off = align256(off + (size_t)NF * 4);
  _Float16* hp     = (_Float16*)((char*)d_ws + off); off = align256(off + (size_t)N * RQ * FOUT * 2);
  (void)ws_size;

  float* outp = (float*)d_out;

  const int T = 256;
  const int packN = RQ * 2 * 32 * 16;

  // zero accumulators (harness poisons d_out/d_ws)
  rgcn_zero_f32<<<(NF + T - 1) / T, T, 0, stream>>>(x, NF);
  rgcn_zero_f32<<<(out_size + T - 1) / T, T, 0, stream>>>(outp, out_size);

  // precision prep: f16 operands, packed B layout
  rgcn_cvt_f16<<<(NF + T - 1) / T, T, 0, stream>>>(feat, feat_h, NF);
  rgcn_pack_w<<<(packN + T - 1) / T, T, 0, stream>>>(W1, Wp1);
  rgcn_pack_w<<<(packN + T - 1) / T, T, 0, stream>>>(W2, Wp2);

  // layer 1
  rgcn_project_wmma<<<1024, 256, 0, stream>>>(feat_h, Wp1, hp, N);
  rgcn_edge_scatter<<<4096, 256, 0, stream>>>(hp, b1, tv, src, dst, et, x, E);
  rgcn_tanh_cvt<<<(NF + T - 1) / T, T, 0, stream>>>(x, xh, NF);

  // layer 2
  rgcn_project_wmma<<<1024, 256, 0, stream>>>(xh, Wp2, hp, N);
  rgcn_edge_scatter<<<4096, 256, 0, stream>>>(hp, b2, tv, src, dst, et, outp, E);
}
